// VisionGNN_13116830122267
// MI455X (gfx1250) — matrix operand order repaired and checked by reading, not compiled, not run on hardware
//
#include <hip/hip_runtime.h>
#include <hip/hip_bf16.h>

typedef __attribute__((ext_vector_type(16))) _Float16 v16h;
typedef __attribute__((ext_vector_type(8)))  _Float16 v8h;
typedef __attribute__((ext_vector_type(8)))  float    v8f;

#define HID 128
#define NGRAPH 64
#define OUTD 10
#define LN_EPS 1e-5f
#define APAD 8   // LDS row pad (halfs): row stride 272B = 17*16B -> conflict-free b128

// ---------------------------------------------------------------------------
// Prep: transpose W [128,128] f32 -> Wt [n][k] f16 so the B-fragment load is
// 16 contiguous halfs (32B) per lane.
// ---------------------------------------------------------------------------
__global__ void cvt_transpose_f16_kernel(const float* __restrict__ W,
                                         _Float16* __restrict__ Wt) {
    int k = blockIdx.x;   // 128
    int n = threadIdx.x;  // 128
    Wt[n * HID + k] = (_Float16)W[k * HID + n];
}

// ---------------------------------------------------------------------------
// Degree / normalization
// ---------------------------------------------------------------------------
__global__ void fill_f32_kernel(float* __restrict__ p, float v, int n) {
    int i = blockIdx.x * blockDim.x + threadIdx.x;
    if (i < n) p[i] = v;
}

__global__ void count_deg_kernel(const int* __restrict__ dst,
                                 float* __restrict__ deg, int E) {
    int e = blockIdx.x * blockDim.x + threadIdx.x;
    if (e < E) atomicAdd(&deg[dst[e]], 1.0f);
}

__global__ void rsqrt_kernel(float* __restrict__ p, int n) {
    int i = blockIdx.x * blockDim.x + threadIdx.x;
    if (i < n) p[i] = rsqrtf(p[i]);  // deg >= 1 always (self loop)
}

// ---------------------------------------------------------------------------
// GEMM: C[M,128] = A[M,128] @ W[128,128], A f32, W pre-transposed f16.
// One block = 256 threads = 8 waves; wave w owns N-tile w (16 cols).
// The 16x128 A tile is staged once per block into LDS as f16 (shared by all
// 8 waves -> 8x less global A traffic and 8x fewer cvt ops than per-wave
// loading), then each lane reads its WMMA fragment as two 16B ds loads.
// M = 50000 is an exact multiple of 16, so EXEC is all-ones for every WMMA.
// ---------------------------------------------------------------------------
__global__ void __launch_bounds__(256)
gemm_wmma_f16_kernel(const float* __restrict__ A,
                     const _Float16* __restrict__ Bt,   // Bt[n*128+k] = W[k,n]
                     float* __restrict__ C, int M) {
    __shared__ alignas(16) _Float16 As[16][HID + APAD];

    // Cooperative stage: 16 rows x 128 cols f32 -> f16 (8 floats per thread)
    {
        int t = threadIdx.x;
        int r = t >> 4;              // 0..15
        int c = (t & 15) * 8;        // 0,8,...,120
        const float* ap = A + ((size_t)blockIdx.x * 16 + r) * HID + c;
        #pragma unroll
        for (int i = 0; i < 8; ++i) As[r][c + i] = (_Float16)ap[i];
    }
    __syncthreads();

    const int lane  = threadIdx.x & 31;
    const int wave  = threadIdx.x >> 5;     // N tile 0..7
    const int half  = lane >> 4;            // lane group 0/1
    const int l     = lane & 15;
    const int colB  = wave * 16 + l;        // B column for this lane

    v8f acc = {};
    #pragma unroll
    for (int kk = 0; kk < HID; kk += 32) {
        // A fragment (contiguous k order, same as B):
        //   lanes 0-15 : h[0..15]=K kk..kk+15
        //   lanes 16-31: h[0..15]=K kk+16..kk+31
        v8h a0 = *reinterpret_cast<const v8h*>(&As[l][kk + half * 16]);
        v8h a1 = *reinterpret_cast<const v8h*>(&As[l][kk + half * 16 + 8]);
        v16h a;
        #pragma unroll
        for (int i = 0; i < 8; ++i) { a[i] = a0[i]; a[8 + i] = a1[i]; }
        // B fragment (contiguous k order, same as A):
        //   lanes 0-15 : h[i]=B[kk+i, n] ; lanes 16-31: h[i]=B[kk+16+i, n]
        v16h b = *reinterpret_cast<const v16h*>(Bt + (size_t)colB * HID + kk + half * 16);

        acc = __builtin_amdgcn_wmma_f32_16x16x32_f16(
            /*neg_a=*/false, a, /*neg_b=*/false, b,
            /*c_mod=*/(short)0, acc, /*reuse_a=*/false, /*reuse_b=*/false);
    }
    // D layout: lane-half 0 -> rows r, lane-half 1 -> rows r+8, N = lane&15.
    float* cp = C + ((size_t)blockIdx.x * 16 + half * 8) * HID + wave * 16 + l;
    #pragma unroll
    for (int r = 0; r < 8; ++r) cp[(size_t)r * HID] = acc[r];
}

// ---------------------------------------------------------------------------
// Aggregation: agg[d] = sum_e norm(e) * h[src(e)]  (+ self loop)
// ---------------------------------------------------------------------------
__global__ void selfloop_init_kernel(const float* __restrict__ h,
                                     const float* __restrict__ dis,
                                     float* __restrict__ agg) {
    int node = blockIdx.x, f = threadIdx.x;       // block = 128
    float d = dis[node];
    size_t i = (size_t)node * HID + f;
    agg[i] = h[i] * d * d;                        // self-loop norm = dis^2
}

__global__ void edge_agg_kernel(const int* __restrict__ src,
                                const int* __restrict__ dst,
                                const float* __restrict__ dis,
                                const float* __restrict__ h,
                                float* __restrict__ agg) {
    int e = blockIdx.x, f = threadIdx.x;          // block = 128
    int s = src[e], d = dst[e];                   // uniform -> scalar loads
    float nrm = dis[s] * dis[d];
    atomicAdd(&agg[(size_t)d * HID + f], h[(size_t)s * HID + f] * nrm);
}

// ---------------------------------------------------------------------------
// Layer 1 finalize: +b1, ReLU, LayerNorm (in place). One block per node.
// ---------------------------------------------------------------------------
__global__ void finalize_ln_kernel(float* __restrict__ h,
                                   const float* __restrict__ b,
                                   const float* __restrict__ g,
                                   const float* __restrict__ be) {
    __shared__ float sm[HID];
    int node = blockIdx.x, f = threadIdx.x;
    size_t i = (size_t)node * HID + f;
    float v = fmaxf(h[i] + b[f], 0.0f);
    sm[f] = v; __syncthreads();
    for (int s = 64; s > 0; s >>= 1) { if (f < s) sm[f] += sm[f + s]; __syncthreads(); }
    float mu = sm[0] * (1.0f / HID);
    __syncthreads();
    float d = v - mu;
    sm[f] = d * d; __syncthreads();
    for (int s = 64; s > 0; s >>= 1) { if (f < s) sm[f] += sm[f + s]; __syncthreads(); }
    float var = sm[0] * (1.0f / HID);
    h[i] = d * rsqrtf(var + LN_EPS) * g[f] + be[f];
}

// ---------------------------------------------------------------------------
// Layer 2 finalize: emb = agg + b2 (written to d_out), r = relu(emb),
// pooled[batch[n]] += r (global add pool).
// ---------------------------------------------------------------------------
__global__ void finalize_emb_pool_kernel(float* __restrict__ emb,
                                         const float* __restrict__ b2,
                                         const int* __restrict__ batch,
                                         float* __restrict__ pooled) {
    int node = blockIdx.x, f = threadIdx.x;
    size_t i = (size_t)node * HID + f;
    float v = emb[i] + b2[f];
    emb[i] = v;
    float r = fmaxf(v, 0.0f);
    atomicAdd(&pooled[(size_t)batch[node] * HID + f], r);
}

// ---------------------------------------------------------------------------
// Head: logits = (pooled@W3+b3)@W4+b4 -> log_softmax. One block per graph.
// Tiny (2 MFLOP) -> plain f32 for logit accuracy.
// ---------------------------------------------------------------------------
__global__ void mlp_head_kernel(const float* __restrict__ pooled,
                                const float* __restrict__ W3,
                                const float* __restrict__ b3,
                                const float* __restrict__ W4,
                                const float* __restrict__ b4,
                                float* __restrict__ out) {
    __shared__ float p[HID];
    __shared__ float hid[HID];
    __shared__ float logits[OUTD];
    int g = blockIdx.x, t = threadIdx.x;
    p[t] = pooled[(size_t)g * HID + t];
    __syncthreads();
    float acc = b3[t];
    #pragma unroll 8
    for (int k = 0; k < HID; ++k) acc = fmaf(p[k], W3[k * HID + t], acc);
    hid[t] = acc;
    __syncthreads();
    if (t < OUTD) {
        float a2 = b4[t];
        for (int k = 0; k < HID; ++k) a2 = fmaf(hid[k], W4[k * OUTD + t], a2);
        logits[t] = a2;
    }
    __syncthreads();
    if (t == 0) {
        float mx = logits[0];
        for (int i = 1; i < OUTD; ++i) mx = fmaxf(mx, logits[i]);
        float s = 0.0f;
        for (int i = 0; i < OUTD; ++i) s += expf(logits[i] - mx);
        float lse = mx + logf(s);
        for (int i = 0; i < OUTD; ++i) out[(size_t)g * OUTD + i] = logits[i] - lse;
    }
}

// ---------------------------------------------------------------------------
extern "C" void kernel_launch(void* const* d_in, const int* in_sizes, int n_in,
                              void* d_out, int out_size, void* d_ws, size_t ws_size,
                              hipStream_t stream) {
    const float* x     = (const float*)d_in[0];
    const int*   eidx  = (const int*)  d_in[1];   // [2,E]: src row then dst row
    const int*   batch = (const int*)  d_in[2];
    const float* W1    = (const float*)d_in[3];
    const float* b1    = (const float*)d_in[4];
    const float* g1    = (const float*)d_in[5];
    const float* be1   = (const float*)d_in[6];
    const float* W2    = (const float*)d_in[7];
    const float* b2    = (const float*)d_in[8];
    const float* W3    = (const float*)d_in[9];
    const float* b3    = (const float*)d_in[10];
    const float* W4    = (const float*)d_in[11];
    const float* b4    = (const float*)d_in[12];

    const int N = in_sizes[0] / HID;              // 50000 (multiple of 16)
    const int E = in_sizes[1] / 2;                // 800000
    const int* src = eidx;
    const int* dst = eidx + E;

    float* emb_out    = (float*)d_out;            // [N,128]
    float* logits_out = emb_out + (size_t)N * HID;// [64,10]

    // Workspace layout
    char* ws = (char*)d_ws;
    size_t off = 0;
    auto carve = [&](size_t bytes) { void* p = ws + off; off += (bytes + 255) & ~size_t(255); return p; };
    float*    dis    = (float*)   carve((size_t)N * 4);
    _Float16* W1t    = (_Float16*)carve((size_t)HID * HID * 2);
    _Float16* W2t    = (_Float16*)carve((size_t)HID * HID * 2);
    float*    pooled = (float*)   carve((size_t)NGRAPH * HID * 4);
    float*    bufA   = (float*)   carve((size_t)N * HID * 4);   // GEMM outputs
    float*    bufB   = (float*)   carve((size_t)N * HID * 4);   // aggregates / h
    (void)ws_size; (void)n_in; (void)out_size;

    // Weight prep (f32 -> transposed f16 for WMMA B fragments)
    cvt_transpose_f16_kernel<<<HID, HID, 0, stream>>>(W1, W1t);
    cvt_transpose_f16_kernel<<<HID, HID, 0, stream>>>(W2, W2t);

    // Degree -> dis = rsqrt(deg), deg initialized to 1 (self loop)
    fill_f32_kernel<<<(N + 255) / 256, 256, 0, stream>>>(dis, 1.0f, N);
    count_deg_kernel<<<(E + 255) / 256, 256, 0, stream>>>(dst, dis, E);
    rsqrt_kernel<<<(N + 255) / 256, 256, 0, stream>>>(dis, N);

    // Layer 1: h1 = x @ W1 (WMMA), aggregate, +b1 / ReLU / LayerNorm
    gemm_wmma_f16_kernel<<<N / 16, 256, 0, stream>>>(x, W1t, bufA, N);
    selfloop_init_kernel<<<N, HID, 0, stream>>>(bufA, dis, bufB);
    edge_agg_kernel<<<E, HID, 0, stream>>>(src, dst, dis, bufA, bufB);
    finalize_ln_kernel<<<N, HID, 0, stream>>>(bufB, b1, g1, be1);

    // Layer 2: h2 = h @ W2 (WMMA), aggregate into d_out emb region
    gemm_wmma_f16_kernel<<<N / 16, 256, 0, stream>>>(bufB, W2t, bufA, N);
    fill_f32_kernel<<<(NGRAPH * HID + 255) / 256, 256, 0, stream>>>(pooled, 0.0f, NGRAPH * HID);
    selfloop_init_kernel<<<N, HID, 0, stream>>>(bufA, dis, emb_out);
    edge_agg_kernel<<<E, HID, 0, stream>>>(src, dst, dis, bufA, emb_out);
    finalize_emb_pool_kernel<<<N, HID, 0, stream>>>(emb_out, b2, batch, pooled);

    // Head + log_softmax
    mlp_head_kernel<<<NGRAPH, HID, 0, stream>>>(pooled, W3, b3, W4, b4, logits_out);
}
